// PredictSparseAttention_84722524881226
// MI455X (gfx1250) — matrix operand order repaired and checked by reading, not compiled, not run on hardware
//
#include <hip/hip_runtime.h>

// ---------------------------------------------------------------------------
// Predicted sparse attention for MI455X (gfx1250, wave32, WMMA + TDM).
// B=4, N=2048, D=1024, K=64, TOPK=64
// ---------------------------------------------------------------------------

#define B_ 4
#define N_ 2048
#define D_ 1024
#define K_ 64
#define TOPK_ 64

typedef __bf16 bf16;
typedef __attribute__((ext_vector_type(16))) __bf16 v16bf;
typedef __attribute__((ext_vector_type(8)))  __bf16 v8bf;
typedef __attribute__((ext_vector_type(4)))  __bf16 v4bf;
typedef __attribute__((ext_vector_type(8)))  float  v8f;
typedef __attribute__((ext_vector_type(4)))  unsigned int u32x4;
typedef __attribute__((ext_vector_type(8)))  int i32x8;
typedef __attribute__((ext_vector_type(4)))  int i32x4;

#if defined(__HIP_DEVICE_COMPILE__) && __has_builtin(__builtin_amdgcn_tensor_load_to_lds)
#define USE_TDM 1
#else
#define USE_TDM 0
#endif

// ---- bf16 helpers (bit-level, avoids reliance on native __bf16 conversions)
static __device__ __forceinline__ bf16 to_bf16(float f) {
    union { float f; unsigned u; } v; v.f = f;
    unsigned r = v.u + 0x7FFFu + ((v.u >> 16) & 1u);   // round-to-nearest-even
    unsigned short h = (unsigned short)(r >> 16);
    return __builtin_bit_cast(bf16, h);
}
static __device__ __forceinline__ float bf2f(bf16 b) {
    unsigned short h = __builtin_bit_cast(unsigned short, b);
    union { unsigned u; float f; } v; v.u = ((unsigned)h) << 16;
    return v.f;
}

// ---------------------------------------------------------------------------
// Kernel 0: fp32 -> bf16 elementwise convert
// ---------------------------------------------------------------------------
__global__ void cvt_f32_bf16(const float* __restrict__ src, bf16* __restrict__ dst, int n) {
    int i = blockIdx.x * blockDim.x + threadIdx.x;
    if (i < n) dst[i] = to_bf16(src[i]);
}

// ---------------------------------------------------------------------------
// Kernel 1: transpose P [D,K] fp32 -> Pt [K,D] bf16 so every GEMM weight is
// [out_features, in_features] row-major (contiguous along the reduction dim).
// ---------------------------------------------------------------------------
__global__ void transpose_P(const float* __restrict__ P, bf16* __restrict__ Pt) {
    int i = blockIdx.x * blockDim.x + threadIdx.x;   // over D_*K_
    if (i < D_ * K_) {
        int d = i / K_, k = i % K_;
        Pt[(size_t)k * D_ + d] = to_bf16(P[i]);
    }
}

// ---------------------------------------------------------------------------
// Tensor Data Mover: issue a 2D tile load (bf16 elements) into LDS.
// D# layout per CDNA5 ISA:
//   group0: count=1 | lds_addr | global_addr(57b) | type=2
//   group1: data_size=1(2B) | tensor_dim0/1 | tile_dim0/1 | tensor_dim0_stride
//   groups 2/3: zero (<=2D tensor)
// All arguments are wave-uniform (derived from blockIdx / k).
// ---------------------------------------------------------------------------
#if USE_TDM
static __device__ __forceinline__ void tdm_load_2d(unsigned lds_off, const void* gptr,
                                                   unsigned tdim0, unsigned tdim1,
                                                   unsigned stride0,
                                                   unsigned tiled0, unsigned tiled1) {
    unsigned long long ga = (unsigned long long)(size_t)gptr;
    u32x4 g0 = { 1u,                                   // count=1, user mode
                 lds_off,                              // LDS byte address
                 (unsigned)ga,                         // global_addr[31:0]
                 (unsigned)((ga >> 32) & 0x01FFFFFFu) | (2u << 30) }; // addr[56:32]|type=2
    i32x8 g1 = { (int)(1u << 16),                      // data_size=1 -> 2 bytes
                 (int)((tdim0 & 0xFFFFu) << 16),       // tensor_dim0[15:0]
                 (int)((tdim0 >> 16) | ((tdim1 & 0xFFFFu) << 16)),
                 (int)((tdim1 >> 16) | (tiled0 << 16)),// tensor_dim1[31:16] | tile_dim0
                 (int)(tiled1 & 0xFFFFu),              // tile_dim1 (tile_dim2=0)
                 (int)stride0,                         // tensor_dim0_stride[31:0]
                 0, 0 };
    i32x4 z4 = { 0, 0, 0, 0 };
#if __clang_major__ >= 23
    i32x8 z8 = { 0, 0, 0, 0, 0, 0, 0, 0 };
    __builtin_amdgcn_tensor_load_to_lds(g0, g1, z4, z4, z8, 0);
#else
    __builtin_amdgcn_tensor_load_to_lds(g0, g1, z4, z4, 0);
#endif
}
#endif

// ---------------------------------------------------------------------------
// Kernel 2: LDS-staged bf16 WMMA GEMM:  C[M,Nn] = A[M,Kd] @ W[Nn,Kd]^T
//
// Block (256 thr = 8 waves) computes a 128x64 macro-tile.
//   Wave w -> 32x32 sub-tile: wm = w>>1 (4 row groups), wn = w&1 (2 col groups),
//   2x2 wmma register blocking (4x v_wmma_f32_16x16x32_bf16 per k-step).
// Per 32-wide k-step, wave 0 issues two TDM tile loads into the next LDS
// buffer (A:128x32, B:64x32, double buffered); compute overlaps the DMA.
// Fragments: A 16x32 and B 32x16 both use k = 16*half + s; C 16x16 f32 per ISA 7.12.2.
// Requires M%128==0, Nn%64==0, Kd%32==0.
// ---------------------------------------------------------------------------
__global__ void gemm_bf16_wmma_lds(const bf16* __restrict__ A,
                                   const bf16* __restrict__ W,
                                   bf16* __restrict__ C,
                                   int M, int Nn, int Kd) {
    __shared__ bf16 As[2][128 * 32];   // [buf][row][k]  8KB each
    __shared__ bf16 Bs[2][64 * 32];    // [buf][col][k]  4KB each

    const int tid  = threadIdx.x;
    const int wave = tid >> 5;
    const int lane = tid & 31;
    const int half = lane >> 4;
    const int l15  = lane & 15;
    const int wm   = wave >> 1;        // 0..3
    const int wn   = wave & 1;         // 0..1

    const int blockM = blockIdx.y * 128;
    const int blockN = blockIdx.x * 64;
    const int nk     = Kd >> 5;

    v8f acc[2][2];
#pragma unroll
    for (int mi = 0; mi < 2; ++mi)
#pragma unroll
        for (int ni = 0; ni < 2; ++ni)
            acc[mi][ni] = (v8f){0.f,0.f,0.f,0.f,0.f,0.f,0.f,0.f};

#if USE_TDM
    // prologue: stage k-block 0 into buffer 0 (one wave drives the TDM)
    if (wave == 0) {
        tdm_load_2d((unsigned)(size_t)&As[0][0], A + (size_t)blockM * Kd,
                    (unsigned)Kd, (unsigned)M, (unsigned)Kd, 32u, 128u);
        tdm_load_2d((unsigned)(size_t)&Bs[0][0], W + (size_t)blockN * Kd,
                    (unsigned)Kd, (unsigned)Nn, (unsigned)Kd, 32u, 64u);
    }
#endif

    for (int kk = 0; kk < nk; ++kk) {
#if USE_TDM
        const int cur = kk & 1;
        if (wave == 0) __builtin_amdgcn_s_wait_tensorcnt(0);
        __syncthreads();               // buffer `cur` ready; prev compute done
        if (wave == 0 && (kk + 1) < nk) {
            const int k1 = (kk + 1) << 5;
            tdm_load_2d((unsigned)(size_t)&As[cur ^ 1][0],
                        A + (size_t)blockM * Kd + k1,
                        (unsigned)Kd, (unsigned)M, (unsigned)Kd, 32u, 128u);
            tdm_load_2d((unsigned)(size_t)&Bs[cur ^ 1][0],
                        W + (size_t)blockN * Kd + k1,
                        (unsigned)Kd, (unsigned)Nn, (unsigned)Kd, 32u, 64u);
        }
#else
        const int cur = 0;
        const int k0 = kk << 5;
        __syncthreads();               // previous compute done before overwrite
        {   // cooperative staging: A 128x32 (16 elem/thread), B 64x32 (128 thr)
            int row = tid >> 1, h = (tid & 1) * 16;
            *(v16bf*)&As[0][row * 32 + h] =
                *(const v16bf*)(A + (size_t)(blockM + row) * Kd + k0 + h);
            if (tid < 128) {
                *(v16bf*)&Bs[0][row * 32 + h] =
                    *(const v16bf*)(W + (size_t)(blockN + row) * Kd + k0 + h);
            }
        }
        __syncthreads();
#endif
        // ---- fragments from LDS + 2x2 wmma
        v16bf afrag[2], bfrag[2];
#pragma unroll
        for (int mi = 0; mi < 2; ++mi) {
            const bf16* ar = &As[cur][((wm * 2 + mi) * 16 + l15) * 32];
            afrag[mi] = *(const v16bf*)(ar + half * 16);
        }
#pragma unroll
        for (int ni = 0; ni < 2; ++ni)
            bfrag[ni] = *(const v16bf*)(&Bs[cur][((wn * 2 + ni) * 16 + l15) * 32 + half * 16]);

#pragma unroll
        for (int mi = 0; mi < 2; ++mi)
#pragma unroll
            for (int ni = 0; ni < 2; ++ni)
                acc[mi][ni] = __builtin_amdgcn_wmma_f32_16x16x32_bf16(
                        false, afrag[mi], false, bfrag[ni],
                        (short)0, acc[mi][ni], false, false);
    }

    // ---- store 32x32 per wave (C/D layout: lane->col l&15, VGPR v->row v+8*half)
#pragma unroll
    for (int mi = 0; mi < 2; ++mi)
#pragma unroll
        for (int ni = 0; ni < 2; ++ni) {
            bf16* Crow = C + (size_t)(blockM + (wm * 2 + mi) * 16 + half * 8) * Nn
                           + blockN + (wn * 2 + ni) * 16 + l15;
#pragma unroll
            for (int v = 0; v < 8; ++v)
                Crow[(size_t)v * Nn] = to_bf16(acc[mi][ni][v]);
        }
}

// ---------------------------------------------------------------------------
// Kernel 3: per-query-row proxy scores + top-64 selection.
// ---------------------------------------------------------------------------
__global__ void topk_kernel(const bf16* __restrict__ tQ,
                            const bf16* __restrict__ tK,
                            int* __restrict__ idx) {
    const int r   = blockIdx.x;          // global row
    const int b   = r >> 11;             // / N_
    const int tid = threadIdx.x;         // 256 threads

    __shared__ float sc[N_];
    __shared__ float q[K_];
    __shared__ float rv[256];
    __shared__ int   ri[256];

    if (tid < K_) q[tid] = bf2f(tQ[(size_t)r * K_ + tid]);
    __syncthreads();

    const bf16* tKbase = tK + (size_t)b * N_ * K_;
    for (int j = tid; j < N_; j += 256) {
        const bf16* krow = tKbase + (size_t)j * K_;
        float s = 0.f;
#pragma unroll 8
        for (int k = 0; k < K_; ++k) s += q[k] * bf2f(krow[k]);
        sc[j] = s;
    }
    __syncthreads();

    for (int it = 0; it < TOPK_; ++it) {
        float best = -__builtin_inff();
        int   bi   = 0x7FFFFFFF;
        for (int j = tid; j < N_; j += 256) {
            float s = sc[j];
            if (s > best || (s == best && j < bi)) { best = s; bi = j; }
        }
        rv[tid] = best; ri[tid] = bi;
        __syncthreads();
        for (int srd = 128; srd > 0; srd >>= 1) {
            if (tid < srd) {
                if (rv[tid + srd] > rv[tid] ||
                    (rv[tid + srd] == rv[tid] && ri[tid + srd] < ri[tid])) {
                    rv[tid] = rv[tid + srd]; ri[tid] = ri[tid + srd];
                }
            }
            __syncthreads();
        }
        if (tid == 0) {
            idx[(size_t)r * TOPK_ + it] = ri[0];
            sc[ri[0]] = -__builtin_inff();
        }
        __syncthreads();
    }
}

// ---------------------------------------------------------------------------
// Kernel 4: sparse attention epilogue (gathered 64-key softmax-weighted sum).
// ---------------------------------------------------------------------------
__global__ void sparse_attn_kernel(const bf16* __restrict__ Q,
                                   const bf16* __restrict__ Kb,
                                   const bf16* __restrict__ Vb,
                                   const int* __restrict__ idx,
                                   float* __restrict__ out) {
    const int r    = blockIdx.x;
    const int b    = r >> 11;
    const int tid  = threadIdx.x;
    const int wave = tid >> 5;
    const int lane = tid & 31;

    __shared__ float qs[D_];
    __shared__ float s[TOPK_];
    __shared__ float a[TOPK_];
    __shared__ int   id[TOPK_];

    for (int d = tid; d < D_; d += 256) qs[d] = bf2f(Q[(size_t)r * D_ + d]);
    if (tid < TOPK_) id[tid] = idx[(size_t)r * TOPK_ + tid];
    __syncthreads();

    const size_t kvbase = (size_t)b * N_ * D_;

    // 8 waves x 8 keys each: wave-parallel 1024-dim dot products
    for (int j = wave * 8; j < wave * 8 + 8; ++j) {
        const bf16* krow = Kb + kvbase + (size_t)id[j] * D_;
        if (j + 1 < wave * 8 + 8)
            __builtin_prefetch(Kb + kvbase + (size_t)id[j + 1] * D_ + lane * 32, 0, 0);
        float p = 0.f;
        for (int d = lane; d < D_; d += 32) p += qs[d] * bf2f(krow[d]);
#pragma unroll
        for (int off = 16; off > 0; off >>= 1) p += __shfl_xor(p, off, 32);
        if (lane == 0) s[j] = p;
    }
    __syncthreads();

    // softmax over the 64 kept scores (computed redundantly, it's tiny)
    float m = -__builtin_inff();
    for (int j = 0; j < TOPK_; ++j) m = fmaxf(m, s[j]);
    float sum = 0.f;
    for (int j = 0; j < TOPK_; ++j) sum += __expf(s[j] - m);
    if (tid < TOPK_) a[tid] = __expf(s[tid] - m) / sum;
    __syncthreads();

    // out[d0..d0+3] = sum_j a_j * V[idx_j][d0..d0+3]
    const int d0 = tid * 4;
    float acc0 = 0.f, acc1 = 0.f, acc2 = 0.f, acc3 = 0.f;
    for (int j = 0; j < TOPK_; ++j) {
        const v4bf vr = *(const v4bf*)(Vb + kvbase + (size_t)id[j] * D_ + d0);
        const float aj = a[j];
        acc0 += aj * bf2f(vr[0]);
        acc1 += aj * bf2f(vr[1]);
        acc2 += aj * bf2f(vr[2]);
        acc3 += aj * bf2f(vr[3]);
    }
    float* orow = out + (size_t)r * D_ + d0;
    orow[0] = acc0; orow[1] = acc1; orow[2] = acc2; orow[3] = acc3;
}

// ---------------------------------------------------------------------------
// Host launcher
// ---------------------------------------------------------------------------
extern "C" void kernel_launch(void* const* d_in, const int* in_sizes, int n_in,
                              void* d_out, int out_size, void* d_ws, size_t ws_size,
                              hipStream_t stream) {
    const float* X   = (const float*)d_in[0];
    const float* P   = (const float*)d_in[1];
    const float* Wqt = (const float*)d_in[2];
    const float* Wkt = (const float*)d_in[3];
    const float* Wq  = (const float*)d_in[4];
    const float* Wk  = (const float*)d_in[5];
    const float* Wv  = (const float*)d_in[6];
    float* out = (float*)d_out;

    const int BN = B_ * N_;

    // workspace carve-up (all offsets 256B aligned)
    char* ws = (char*)d_ws;
    size_t off = 0;
    auto take = [&](size_t bytes) -> void* {
        void* p = ws + off;
        off = (off + bytes + 255) & ~(size_t)255;
        return p;
    };
    bf16* Xb   = (bf16*)take((size_t)BN * D_ * 2);    // X in bf16
    bf16* Ptb  = (bf16*)take((size_t)K_ * D_ * 2);    // P^T in bf16  [K,D]
    bf16* Wqtb = (bf16*)take((size_t)K_ * K_ * 2);
    bf16* Wktb = (bf16*)take((size_t)K_ * K_ * 2);
    bf16* Wqb  = (bf16*)take((size_t)D_ * D_ * 2);
    bf16* Wkb  = (bf16*)take((size_t)D_ * D_ * 2);
    bf16* Wvb  = (bf16*)take((size_t)D_ * D_ * 2);
    bf16* Xpb  = (bf16*)take((size_t)BN * K_ * 2);    // X @ P
    bf16* tQb  = (bf16*)take((size_t)BN * K_ * 2);
    bf16* tKb  = (bf16*)take((size_t)BN * K_ * 2);
    bf16* Qb   = (bf16*)take((size_t)BN * D_ * 2);
    bf16* Kbf  = (bf16*)take((size_t)BN * D_ * 2);
    bf16* Vbf  = (bf16*)take((size_t)BN * D_ * 2);
    int*  idx  = (int*)take((size_t)BN * TOPK_ * 4);
    (void)ws_size; (void)n_in; (void)in_sizes; (void)out_size;

    auto cvt = [&](const float* s, bf16* d, int n) {
        cvt_f32_bf16<<<(n + 255) / 256, 256, 0, stream>>>(s, d, n);
    };

    // 1) precision conversion + P transpose
    cvt(X,   Xb,   BN * D_);
    cvt(Wqt, Wqtb, K_ * K_);
    cvt(Wkt, Wktb, K_ * K_);
    cvt(Wq,  Wqb,  D_ * D_);
    cvt(Wk,  Wkb,  D_ * D_);
    cvt(Wv,  Wvb,  D_ * D_);
    transpose_P<<<(D_ * K_ + 255) / 256, 256, 0, stream>>>(P, Ptb);

    // 2) LDS-staged WMMA GEMMs: block macro-tile 128x64
    auto gemm = [&](const bf16* A, const bf16* W, bf16* C, int M, int Nn, int Kd) {
        dim3 grid(Nn / 64, M / 128);
        gemm_bf16_wmma_lds<<<grid, 256, 0, stream>>>(A, W, C, M, Nn, Kd);
    };
    gemm(Xb,  Ptb,  Xpb, BN, K_, D_);   // Xp = X @ P
    gemm(Xpb, Wqtb, tQb, BN, K_, K_);   // tQ = Xp @ Wq_tilde^T
    gemm(Xpb, Wktb, tKb, BN, K_, K_);   // tK = Xp @ Wk_tilde^T
    gemm(Xb,  Wqb,  Qb,  BN, D_, D_);   // Q  = X @ Wq^T
    gemm(Xb,  Wkb,  Kbf, BN, D_, D_);   // K  = X @ Wk^T
    gemm(Xb,  Wvb,  Vbf, BN, D_, D_);   // V  = X @ Wv^T

    // 3) proxy-score top-64 per query row
    topk_kernel<<<BN, 256, 0, stream>>>(tQb, tKb, idx);

    // 4) gathered sparse attention
    sparse_attn_kernel<<<BN, 256, 0, stream>>>(Qb, Kbf, Vbf, idx, out);
}
